// LSTM_Background_25993142076015
// MI455X (gfx1250) — matrix operand order repaired and checked by reading, not compiled, not run on hardware
//
#include <hip/hip_runtime.h>
#include <hip/hip_bf16.h>
#include <math.h>
#include <stdint.h>

typedef __attribute__((ext_vector_type(16))) __bf16   v16bf;
typedef __attribute__((ext_vector_type(8)))  float    v8f;
typedef __attribute__((ext_vector_type(4)))  uint32_t u32x4;
typedef __attribute__((ext_vector_type(8)))  uint32_t u32x8;

#define HDIM  128
#define KNB   32
#define NPAIR 6
#define OUTD  64

// ---------------------------------------------------------------------------
// Async global -> LDS copy helpers (ASYNCcnt path, §10 / §15.18).
// lds_off: byte offset of destination within LDS; ga: 64-bit global address.
// ---------------------------------------------------------------------------
__device__ __forceinline__ void async_copy_b128(uint32_t lds_off, uint64_t ga) {
  asm volatile("global_load_async_to_lds_b128 %0, %1, off"
               :: "v"(lds_off), "v"(ga) : "memory");
}
__device__ __forceinline__ void async_wait0() {
  asm volatile("s_wait_asynccnt 0x0" ::: "memory");
}
__device__ __forceinline__ uint32_t lds_addr_of(const void* p) {
  return (uint32_t)(size_t)p;   // generic LDS pointer: low 32 bits = LDS offset
}

// ---------------------------------------------------------------------------
// TDM: load a 2-D tile (dim1 rows x dim0 cols, 2-byte elements, row-major,
// stride == dim0) from global into LDS. Minimal D# per ISA §8; 2-group form
// (VADDR2/3 NULL) is valid for tensors up to 2-D. Issued per-wave; EXEC
// ignored by TDM. Completion tracked with TENSORcnt.
// ---------------------------------------------------------------------------
__device__ __forceinline__ void tdm_load_2d_bf16(const void* gptr, uint32_t lds_off,
                                                 uint32_t dim0, uint32_t dim1) {
  const uint64_t ga = (uint64_t)(size_t)gptr;
  u32x4 g0;
  g0[0] = 1u;                                            // count=1, user desc
  g0[1] = lds_off;                                       // lds_addr
  g0[2] = (uint32_t)ga;                                  // global_addr[31:0]
  g0[3] = (uint32_t)((ga >> 32) & 0x01FFFFFFu)           // global_addr[56:32]
        | (2u << 30);                                    // type = 2 (image)
  u32x8 g1;
  g1[0] = (1u << 16);                                    // data_size=1 -> 2 bytes
  g1[1] = (dim0 & 0xFFFFu) << 16;                        // tensor_dim0[15:0]
  g1[2] = ((dim0 >> 16) & 0xFFFFu)                       // tensor_dim0[31:16]
        | ((dim1 & 0xFFFFu) << 16);                      // tensor_dim1[15:0]
  g1[3] = ((dim1 >> 16) & 0xFFFFu)                       // tensor_dim1[31:16]
        | ((dim0 & 0xFFFFu) << 16);                      // tile_dim0
  g1[4] = (dim1 & 0xFFFFu);                              // tile_dim1 (tile_dim2=0)
  g1[5] = dim0;                                          // tensor_dim0_stride[31:0]
  g1[6] = 0u;                                            // stride hi / dim1_stride lo
  g1[7] = 0u;
  asm volatile("tensor_load_to_lds %0, %1" :: "s"(g0), "s"(g1) : "memory");
}
__device__ __forceinline__ void tensor_wait0() {
#if defined(__has_builtin)
#if __has_builtin(__builtin_amdgcn_s_wait_tensorcnt)
  __builtin_amdgcn_s_wait_tensorcnt(0);
#else
  asm volatile("s_wait_tensorcnt 0x0" ::: "memory");
#endif
#else
  asm volatile("s_wait_tensorcnt 0x0" ::: "memory");
#endif
}

// ---------------------------------------------------------------------------
// Top-32 per row (indices sorted ascending). One block per row; the 40KB row
// is staged into LDS with async b128 copies, then 32 rounds of block arg-max.
// ---------------------------------------------------------------------------
__global__ void topk_rows_kernel(const float* __restrict__ sim, int n,
                                 int* __restrict__ out_map) {
  __shared__ __attribute__((aligned(16))) float row[10016];
  __shared__ float red_v[256];
  __shared__ int   red_i[256];
  __shared__ int   sel[KNB];
  const int r = blockIdx.x;
  const float* src = sim + (size_t)r * n;

  {  // async stage: n*4 bytes in 16-byte chunks (ASYNCcnt path)
    const uint32_t base = lds_addr_of(row);
    const int nchunk = (n * 4) >> 4;
    for (int c = threadIdx.x; c < nchunk; c += blockDim.x)
      async_copy_b128(base + (uint32_t)c * 16u,
                      (uint64_t)(size_t)((const char*)src + (size_t)c * 16u));
    for (int i = (nchunk << 2) + threadIdx.x; i < n; i += blockDim.x)
      row[i] = src[i];  // tail (n % 4), none for n=10000
    async_wait0();
  }
  __syncthreads();

  for (int k = 0; k < KNB; ++k) {
    float bv = -INFINITY; int bidx = 0x7fffffff;
    for (int i = threadIdx.x; i < n; i += blockDim.x) {
      float v = row[i];
      if (v > bv) { bv = v; bidx = i; }
    }
    red_v[threadIdx.x] = bv; red_i[threadIdx.x] = bidx;
    __syncthreads();
    for (int s = 128; s > 0; s >>= 1) {
      if ((int)threadIdx.x < s) {
        float ov = red_v[threadIdx.x + s]; int oi = red_i[threadIdx.x + s];
        if (ov > red_v[threadIdx.x] ||
            (ov == red_v[threadIdx.x] && oi < red_i[threadIdx.x])) {
          red_v[threadIdx.x] = ov; red_i[threadIdx.x] = oi;
        }
      }
      __syncthreads();
    }
    if (threadIdx.x == 0) { sel[k] = red_i[0]; row[red_i[0]] = -INFINITY; }
    __syncthreads();
  }
  if (threadIdx.x == 0) {  // sort ascending (jnp.sort of top-k indices)
    for (int i = 1; i < KNB; ++i) {
      int v = sel[i]; int j = i - 1;
      while (j >= 0 && sel[j] > v) { sel[j + 1] = sel[j]; --j; }
      sel[j + 1] = v;
    }
  }
  __syncthreads();
  if (threadIdx.x < KNB) out_map[(size_t)r * KNB + threadIdx.x] = sel[threadIdx.x];
}

// ---------------------------------------------------------------------------
__global__ void cvt_bf16_kernel(const float* __restrict__ in,
                                __bf16* __restrict__ out, int n) {
  int i = blockIdx.x * blockDim.x + threadIdx.x;
  if (i < n) out[i] = (__bf16)in[i];
}

// ---------------------------------------------------------------------------
// Per-sample features a,b,c,d; c/d = 32x7 conv over gathered neighbor tiles.
// Neighbor tiles (32x128 f32) staged with async b128 copies.
// Block = 128 threads (one per H channel), grid = B.
// ---------------------------------------------------------------------------
__global__ void feature_kernel(const int* __restrict__ data,
                               const float* __restrict__ user_emb,
                               const float* __restrict__ service_emb,
                               const int* __restrict__ u_map,
                               const int* __restrict__ s_map,
                               const float* __restrict__ cnn_w,
                               const float* __restrict__ cnn_b,
                               const float* __restrict__ scnn_w,
                               const float* __restrict__ scnn_b,
                               float* __restrict__ sp_f32,
                               __bf16* __restrict__ sp_bf) {
  __shared__ __attribute__((aligned(16))) float nb[KNB][HDIM];
  __shared__ float wsh[KNB * 7];
  __shared__ int   um[KNB];
  const int b = blockIdx.x;
  const int t = threadIdx.x;           // 0..127 (H channel)
  const int uid = data[b * 3 + 1];
  const int sid = data[b * 3 + 2];
  const uint32_t nb_base = lds_addr_of(&nb[0][0]);

  // ---- user neighborhood conv -> c ----
  if (t < KNB) um[t] = u_map[(size_t)uid * KNB + t];
  for (int i = t; i < KNB * 7; i += HDIM) wsh[i] = cnn_w[i];
  __syncthreads();
  for (int idx = t; idx < KNB * (HDIM / 4); idx += HDIM) {  // 16B chunks
    int j = idx >> 5, ch = idx & 31;
    async_copy_b128(nb_base + (uint32_t)(j * HDIM + ch * 4) * 4u,
                    (uint64_t)(size_t)(user_emb + (size_t)um[j] * HDIM + ch * 4));
  }
  async_wait0();
  __syncthreads();
  float c = cnn_b[0];
  for (int j = 0; j < KNB; ++j) {
    #pragma unroll
    for (int kw = 0; kw < 7; ++kw) {
      int w = t + kw - 3;
      float xv = (w >= 0 && w < HDIM) ? nb[j][w] : 0.0f;
      c += xv * wsh[j * 7 + kw];
    }
  }
  __syncthreads();

  // ---- service neighborhood conv -> d ----
  if (t < KNB) um[t] = s_map[(size_t)sid * KNB + t];
  for (int i = t; i < KNB * 7; i += HDIM) wsh[i] = scnn_w[i];
  __syncthreads();
  for (int idx = t; idx < KNB * (HDIM / 4); idx += HDIM) {
    int j = idx >> 5, ch = idx & 31;
    async_copy_b128(nb_base + (uint32_t)(j * HDIM + ch * 4) * 4u,
                    (uint64_t)(size_t)(service_emb + (size_t)um[j] * HDIM + ch * 4));
  }
  async_wait0();
  __syncthreads();
  float d = scnn_b[0];
  for (int j = 0; j < KNB; ++j) {
    #pragma unroll
    for (int kw = 0; kw < 7; ++kw) {
      int w = t + kw - 3;
      float xv = (w >= 0 && w < HDIM) ? nb[j][w] : 0.0f;
      d += xv * wsh[j * 7 + kw];
    }
  }

  const float a  = user_emb[(size_t)uid * HDIM + t];
  const float bv = service_emb[(size_t)sid * HDIM + t];
  float f[4] = {a, bv, c, d};
  const int ii[NPAIR] = {0, 0, 0, 1, 1, 2};
  const int jj[NPAIR] = {1, 2, 3, 2, 3, 3};
  const size_t base = (size_t)b * NPAIR * HDIM + t;
  #pragma unroll
  for (int p = 0; p < NPAIR; ++p) {
    float s = f[ii[p]] + f[jj[p]];
    sp_f32[base + (size_t)p * HDIM] = s;
    sp_bf [base + (size_t)p * HDIM] = (__bf16)s;
  }
}

// ---------------------------------------------------------------------------
// Gate GEMMs: forget = sigmoid(sp@wf.T+bf), t = sp@wi.T+bi,
// tempt = relu(forget*t + (1-forget)*sp); norms = ||tempt|| per row.
// wf/wi (bf16, 32KB each) staged into LDS via the Tensor Data Mover; waves
// then run 8 n-tiles x 4 k-chunks of v_wmma_f32_16x16x32_bf16 with B-frags
// served from LDS. 8 waves/block, 16 rows/wave.
// ---------------------------------------------------------------------------
__global__ void gate_wmma_kernel(const __bf16* __restrict__ sp_bf,
                                 const float*  __restrict__ sp_f32,
                                 const __bf16* __restrict__ wf_bf,
                                 const __bf16* __restrict__ wi_bf,
                                 const float*  __restrict__ bf_bias,
                                 const float*  __restrict__ bi_bias,
                                 float* __restrict__ tempt,
                                 float* __restrict__ norms) {
  __shared__ __attribute__((aligned(16))) __bf16 lwf[HDIM * HDIM];
  __shared__ __attribute__((aligned(16))) __bf16 lwi[HDIM * HDIM];
  if (threadIdx.x < 32) {        // wave 0 issues both TDM tile loads
    tdm_load_2d_bf16(wf_bf, lds_addr_of(lwf), HDIM, HDIM);
    tdm_load_2d_bf16(wi_bf, lds_addr_of(lwi), HDIM, HDIM);
    tensor_wait0();
  }
  __syncthreads();

  const int wave = threadIdx.x >> 5;
  const int lane = threadIdx.x & 31;
  const int g  = lane >> 4;        // lane half (K-split for A/B; M-split for D)
  const int lr = lane & 15;
  const int m0 = blockIdx.x * 128 + wave * 16;
  const __bf16* arow = sp_bf + (size_t)(m0 + lr) * HDIM;
  __builtin_prefetch(arow + 16 * HDIM, 0, 0);

  float nrm[8];
  #pragma unroll
  for (int r = 0; r < 8; ++r) nrm[r] = 0.0f;

  for (int nt = 0; nt < 8; ++nt) {
    const int n0 = nt * 16;
    const int wbase = (n0 + lr) * HDIM;
    v8f accf = {};
    v8f acct = {};
    #pragma unroll
    for (int kc = 0; kc < 4; ++kc) {
      const int k0 = kc * 32;
      v16bf a, bF, bI;
      #pragma unroll
      for (int i = 0; i < 8; ++i) {            // A: K split by lane half (contiguous, matches B)
        a[i]     = arow[k0 + g * 16 + i];
        a[8 + i] = arow[k0 + g * 16 + 8 + i];
      }
      #pragma unroll
      for (int i = 0; i < 16; ++i) {           // B = W^T, K split by lane half
        bF[i] = lwf[wbase + k0 + g * 16 + i];
        bI[i] = lwi[wbase + k0 + g * 16 + i];
      }
      accf = __builtin_amdgcn_wmma_f32_16x16x32_bf16(false, a, false, bF,
                                                     (short)0, accf, false, false);
      acct = __builtin_amdgcn_wmma_f32_16x16x32_bf16(false, a, false, bI,
                                                     (short)0, acct, false, false);
    }
    const int   col = n0 + lr;
    const float bfv = bf_bias[col];
    const float biv = bi_bias[col];
    #pragma unroll
    for (int r = 0; r < 8; ++r) {              // D layout: M = r + 8*g, N = lr
      const int row = m0 + g * 8 + r;
      float fgate = 1.0f / (1.0f + __expf(-(accf[r] + bfv)));
      float tv    = acct[r] + biv;
      float spv   = sp_f32[(size_t)row * HDIM + col];
      float te    = fgate * tv + (1.0f - fgate) * spv;
      te = te > 0.0f ? te : 0.0f;
      tempt[(size_t)row * HDIM + col] = te;
      nrm[r] += te * te;
    }
  }
  #pragma unroll
  for (int r = 0; r < 8; ++r) {                // reduce over 16 lanes (same rows)
    for (int o = 8; o >= 1; o >>= 1) nrm[r] += __shfl_xor(nrm[r], o, 16);
  }
  if (lr == 0) {
    #pragma unroll
    for (int r = 0; r < 8; ++r) norms[m0 + g * 8 + r] = sqrtf(nrm[r]);
  }
}

// ---------------------------------------------------------------------------
// Top-4 of the 6 pair norms per sample; gather rows into x (B x 512) bf16.
// ---------------------------------------------------------------------------
__global__ void select_kernel(const float* __restrict__ norms,
                              const float* __restrict__ tempt,
                              __bf16* __restrict__ x_bf) {
  const int b = blockIdx.x;
  const int t = threadIdx.x;  // 0..127
  float nv[NPAIR];
  #pragma unroll
  for (int p = 0; p < NPAIR; ++p) nv[p] = norms[(size_t)b * NPAIR + p];
  int  idx[4];
  bool used[NPAIR] = {false, false, false, false, false, false};
  #pragma unroll
  for (int s = 0; s < 4; ++s) {
    float best = -INFINITY; int bp = 0;
    #pragma unroll
    for (int p = 0; p < NPAIR; ++p)
      if (!used[p] && nv[p] > best) { best = nv[p]; bp = p; }
    used[bp] = true; idx[s] = bp;
  }
  #pragma unroll
  for (int s = 0; s < 4; ++s)
    x_bf[(size_t)b * 512 + s * HDIM + t] =
        (__bf16)tempt[((size_t)b * NPAIR + idx[s]) * HDIM + t];
}

// ---------------------------------------------------------------------------
// Final FC: (B x 512) @ (512 x 64) + bias, f32 out. fc_w (64KB bf16) staged
// into LDS via TDM; 16 K-chunks of bf16 WMMA per 16-col tile.
// ---------------------------------------------------------------------------
__global__ void fc_wmma_kernel(const __bf16* __restrict__ x_bf,
                               const __bf16* __restrict__ fcw_bf,
                               const float*  __restrict__ fc_b,
                               float* __restrict__ out) {
  __shared__ __attribute__((aligned(16))) __bf16 lw[OUTD * 4 * HDIM];
  if (threadIdx.x < 32) {
    tdm_load_2d_bf16(fcw_bf, lds_addr_of(lw), 4 * HDIM, OUTD);
    tensor_wait0();
  }
  __syncthreads();

  const int wave = threadIdx.x >> 5;
  const int lane = threadIdx.x & 31;
  const int g  = lane >> 4;
  const int lr = lane & 15;
  const int m0 = blockIdx.x * 128 + wave * 16;
  const __bf16* arow = x_bf + (size_t)(m0 + lr) * 512;
  __builtin_prefetch(arow + 16 * 512, 0, 0);

  for (int nt = 0; nt < 4; ++nt) {
    const int n0 = nt * 16;
    const int wbase = (n0 + lr) * 512;
    v8f acc = {};
    #pragma unroll
    for (int kc = 0; kc < 16; ++kc) {
      const int k0 = kc * 32;
      v16bf a, bW;
      #pragma unroll
      for (int i = 0; i < 8; ++i) {
        a[i]     = arow[k0 + g * 16 + i];
        a[8 + i] = arow[k0 + g * 16 + 8 + i];
      }
      #pragma unroll
      for (int i = 0; i < 16; ++i) bW[i] = lw[wbase + k0 + g * 16 + i];
      acc = __builtin_amdgcn_wmma_f32_16x16x32_bf16(false, a, false, bW,
                                                    (short)0, acc, false, false);
    }
    const int   col  = n0 + lr;
    const float bias = fc_b[col];
    #pragma unroll
    for (int r = 0; r < 8; ++r)
      out[(size_t)(m0 + g * 8 + r) * OUTD + col] = acc[r] + bias;
  }
}

// ---------------------------------------------------------------------------
extern "C" void kernel_launch(void* const* d_in, const int* in_sizes, int n_in,
                              void* d_out, int out_size, void* d_ws, size_t ws_size,
                              hipStream_t stream) {
  (void)n_in; (void)out_size; (void)ws_size;
  const int*   data        = (const int*)  d_in[0];
  const float* user_sim    = (const float*)d_in[1];
  const float* service_sim = (const float*)d_in[2];
  const float* user_emb    = (const float*)d_in[3];
  const float* service_emb = (const float*)d_in[4];
  const float* cnn_w  = (const float*)d_in[5];
  const float* cnn_b  = (const float*)d_in[6];
  const float* scnn_w = (const float*)d_in[7];
  const float* scnn_b = (const float*)d_in[8];
  const float* wi     = (const float*)d_in[9];
  const float* bi     = (const float*)d_in[10];
  const float* wf     = (const float*)d_in[11];
  const float* bfb    = (const float*)d_in[12];
  const float* fc_w   = (const float*)d_in[13];
  const float* fc_b   = (const float*)d_in[14];
  float* out = (float*)d_out;

  const int B = in_sizes[0] / 3;
  const int U = in_sizes[3] / HDIM;
  const int S = in_sizes[4] / HDIM;
  const int M = B * NPAIR;   // 6B rows of 128

  size_t off = 0;
  auto alloc = [&](size_t bytes) -> void* {
    off = (off + 255) & ~(size_t)255;
    void* p = (char*)d_ws + off;
    off += bytes;
    return p;
  };
  int*    u_map  = (int*)   alloc((size_t)U * KNB * 4);
  int*    s_map  = (int*)   alloc((size_t)S * KNB * 4);
  __bf16* wf_bf  = (__bf16*)alloc((size_t)HDIM * HDIM * 2);
  __bf16* wi_bf  = (__bf16*)alloc((size_t)HDIM * HDIM * 2);
  __bf16* fcw_bf = (__bf16*)alloc((size_t)OUTD * 4 * HDIM * 2);
  float*  sp_f32 = (float*) alloc((size_t)M * HDIM * 4);
  __bf16* sp_bf  = (__bf16*)alloc((size_t)M * HDIM * 2);
  float*  tempt  = (float*) alloc((size_t)M * HDIM * 4);
  float*  norms  = (float*) alloc((size_t)M * 4);
  __bf16* x_bf   = (__bf16*)alloc((size_t)B * 4 * HDIM * 2);

  topk_rows_kernel<<<U, 256, 0, stream>>>(user_sim, U, u_map);
  topk_rows_kernel<<<S, 256, 0, stream>>>(service_sim, S, s_map);
  cvt_bf16_kernel<<<(HDIM * HDIM + 255) / 256, 256, 0, stream>>>(wf, wf_bf, HDIM * HDIM);
  cvt_bf16_kernel<<<(HDIM * HDIM + 255) / 256, 256, 0, stream>>>(wi, wi_bf, HDIM * HDIM);
  cvt_bf16_kernel<<<(OUTD * 4 * HDIM + 255) / 256, 256, 0, stream>>>(fc_w, fcw_bf, OUTD * 4 * HDIM);
  feature_kernel<<<B, HDIM, 0, stream>>>(data, user_emb, service_emb, u_map, s_map,
                                         cnn_w, cnn_b, scnn_w, scnn_b, sp_f32, sp_bf);
  gate_wmma_kernel<<<M / 128, 256, 0, stream>>>(sp_bf, sp_f32, wf_bf, wi_bf,
                                                bfb, bi, tempt, norms);
  select_kernel<<<B, HDIM, 0, stream>>>(norms, tempt, x_bf);
  fc_wmma_kernel<<<B / 128, 256, 0, stream>>>(x_bf, fcw_bf, fc_b, out);
}
